// TubeletEmbedding_87771951661498
// MI455X (gfx1250) — matrix operand order repaired and checked by reading, not compiled, not run on hardware
//
#include <hip/hip_runtime.h>

// ---- CDNA5 WMMA vector types ----
typedef __attribute__((ext_vector_type(16))) __bf16 v16bf;
typedef __attribute__((ext_vector_type(8)))  __bf16 v8bf;
typedef __attribute__((ext_vector_type(8)))  float  v8f;

// Problem constants (from reference)
#define BATCH   64
#define TSTACK  16
#define IMG     84
#define EMBED   1024
#define KDIM    98          // TUBE * 7*7
#define KP      128         // K padded to multiple of 32 for bf16 WMMA
#define NTOK    1152        // 8 * 144 tokens
#define NROWS   (BATCH*NTOK)// 73728 output rows

// Tiling
#define MT      64          // rows per workgroup
#define NT      128         // cols per workgroup
#define LDK     136         // LDS k-stride in bf16 elems (68 dwords -> conflict-free frags)
#define WGSIZE  256         // 8 waves

#define SMEM_BYTES ((2*MT + 2*NT) * LDK * 2)   /* hi+lo for A and B = 104448 B */

__global__ __launch_bounds__(WGSIZE)
void tubelet_embed_wmma_bf16x3(const float* __restrict__ video,
                               const float* __restrict__ Wm,
                               const float* __restrict__ bias,
                               const float* __restrict__ pos,
                               float* __restrict__ out)
{
    extern __shared__ __bf16 smem[];
    __bf16* sAh = smem;                  // [MT][LDK]
    __bf16* sAl = sAh + MT * LDK;        // [MT][LDK]
    __bf16* sBh = sAl + MT * LDK;        // [NT][LDK]
    __bf16* sBl = sBh + NT * LDK;        // [NT][LDK]

    const int tid   = threadIdx.x;
    const int ntile = blockIdx.x;        // 0..7   (EMBED/NT)
    const int mtile = blockIdx.y;        // 0..1151 (NROWS/MT)

    // ---------------- A tile: gather tubelets from video, split fp32 -> bf16 hi/lo ----
    for (int idx = tid; idx < MT * KDIM; idx += WGSIZE) {
        int r   = idx / KDIM;
        int k   = idx - r * KDIM;
        int row = mtile * MT + r;
        int bb  = row / NTOK;
        int n   = row - bb * NTOK;
        int d   = n / 144;               // temporal tubelet index
        int s   = n - d * 144;           // spatial token
        int gh  = s / 12;
        int gw  = s - gh * 12;
        int ts  = k / 49;                // frame within tubelet
        int k2  = k - ts * 49;
        int pr  = k2 / 7;
        int pc  = k2 - pr * 7;
        float v = video[(((bb * TSTACK) + d * 2 + ts) * IMG + gh * 7 + pr) * IMG
                        + gw * 7 + pc];
        __bf16 hi = (__bf16)v;
        __bf16 lo = (__bf16)(v - (float)hi);
        sAh[r * LDK + k] = hi;
        sAl[r * LDK + k] = lo;
    }
    for (int idx = tid; idx < MT * (KP - KDIM); idx += WGSIZE) {  // zero K padding
        int r = idx / (KP - KDIM);
        int k = KDIM + (idx - r * (KP - KDIM));
        sAh[r * LDK + k] = (__bf16)0.0f;
        sAl[r * LDK + k] = (__bf16)0.0f;
    }
    // ---------------- B tile: W[e,k], stored [n][k] so B-frags are contiguous in K ----
    for (int idx = tid; idx < NT * KDIM; idx += WGSIZE) {
        int c = idx / KDIM;
        int k = idx - c * KDIM;
        float v = Wm[(ntile * NT + c) * KDIM + k];
        __bf16 hi = (__bf16)v;
        __bf16 lo = (__bf16)(v - (float)hi);
        sBh[c * LDK + k] = hi;
        sBl[c * LDK + k] = lo;
    }
    for (int idx = tid; idx < NT * (KP - KDIM); idx += WGSIZE) {
        int c = idx / (KP - KDIM);
        int k = KDIM + (idx - c * (KP - KDIM));
        sBh[c * LDK + k] = (__bf16)0.0f;
        sBl[c * LDK + k] = (__bf16)0.0f;
    }
    __syncthreads();

    // ---------------- WMMA compute: each wave does a 16x64 strip -------------------
    const int w    = tid >> 5;
    const int lane = tid & 31;
    const int h    = lane >> 4;          // lane half (ISA A/B layout)
    const int l16  = lane & 15;
    const int mrow = (w & 3) * 16;       // wave row offset in tile
    const int ncol = (w >> 2) * 64;      // wave col offset in tile

    v8f acc[4] = {};

    for (int k0 = 0; k0 < KP; k0 += 32) {
        // A fragments (shared across the 4 N sub-tiles)
        const __bf16* ahp = &sAh[(mrow + l16) * LDK + k0 + 8 * h];
        const __bf16* alp = &sAl[(mrow + l16) * LDK + k0 + 8 * h];
        v8bf ah0 = *(const v8bf*)ahp;
        v8bf ah1 = *(const v8bf*)(ahp + 16);
        v8bf al0 = *(const v8bf*)alp;
        v8bf al1 = *(const v8bf*)(alp + 16);
        v16bf aH = __builtin_shufflevector(ah0, ah1, 0,1,2,3,4,5,6,7,8,9,10,11,12,13,14,15);
        v16bf aL = __builtin_shufflevector(al0, al1, 0,1,2,3,4,5,6,7,8,9,10,11,12,13,14,15);

        #pragma unroll
        for (int sub = 0; sub < 4; ++sub) {
            const __bf16* bhp = &sBh[(ncol + sub * 16 + l16) * LDK + k0 + 8 * h];
            const __bf16* blp = &sBl[(ncol + sub * 16 + l16) * LDK + k0 + 8 * h];
            v8bf bh0 = *(const v8bf*)bhp;
            v8bf bh1 = *(const v8bf*)(bhp + 16);
            v8bf bl0 = *(const v8bf*)blp;
            v8bf bl1 = *(const v8bf*)(blp + 16);
            v16bf bH = __builtin_shufflevector(bh0, bh1, 0,1,2,3,4,5,6,7,8,9,10,11,12,13,14,15);
            v16bf bL = __builtin_shufflevector(bl0, bl1, 0,1,2,3,4,5,6,7,8,9,10,11,12,13,14,15);

            // split-bf16 fp32 emulation: hi*hi + hi*lo + lo*hi
            acc[sub] = __builtin_amdgcn_wmma_f32_16x16x32_bf16(
                false, aH, false, bH, (short)0, acc[sub], false, false);
            acc[sub] = __builtin_amdgcn_wmma_f32_16x16x32_bf16(
                false, aH, false, bL, (short)0, acc[sub], false, false);
            acc[sub] = __builtin_amdgcn_wmma_f32_16x16x32_bf16(
                false, aL, false, bH, (short)0, acc[sub], false, false);
        }
    }

    // ---------------- epilogue: + bias + pos_embed, store ---------------------------
    // C/D layout: VGPR v, lanes 0-15 -> M=v, lanes 16-31 -> M=v+8; N = lane&15
    const int n0    = (mtile % (NTOK / MT)) * MT;       // token base of this row tile
    const int rbase = mtile * MT + mrow + 8 * h;        // global row for acc elem v=0
    const int nbase = n0 + mrow + 8 * h;                // token index for acc elem v=0

    #pragma unroll
    for (int sub = 0; sub < 4; ++sub) {
        const int col = ntile * NT + ncol + sub * 16 + l16;
        const float bv = bias[col];
        const float* pc = pos + (size_t)nbase * EMBED + col;
        float*       oc = out + (size_t)rbase * EMBED + col;
        #pragma unroll
        for (int v = 0; v < 8; ++v) {
            oc[(size_t)v * EMBED] = acc[sub][v] + bv + pc[(size_t)v * EMBED];
        }
    }
}

extern "C" void kernel_launch(void* const* d_in, const int* in_sizes, int n_in,
                              void* d_out, int out_size, void* d_ws, size_t ws_size,
                              hipStream_t stream) {
    (void)in_sizes; (void)n_in; (void)d_ws; (void)ws_size; (void)out_size;
    const float* video = (const float*)d_in[0];   // [64,16,1,84,84]
    const float* Wm    = (const float*)d_in[1];   // [1024,98]
    const float* bias  = (const float*)d_in[2];   // [1024]
    const float* pos   = (const float*)d_in[3];   // [1,1152,1024]
    float* out = (float*)d_out;                   // [64,1152,1024]

    dim3 grid(EMBED / NT, NROWS / MT);            // (8, 1152)
    dim3 block(WGSIZE);
    tubelet_embed_wmma_bf16x3<<<grid, block, SMEM_BYTES, stream>>>(
        video, Wm, bias, pos, out);
}
